// SparseMMRecBlock_29154238005387
// MI455X (gfx1250) — matrix operand order repaired and checked by reading, not compiled, not run on hardware
//
#include <hip/hip_runtime.h>
#include <hip/hip_bf16.h>
#include <math.h>

// ---------------- problem constants (fixed by the reference) ----------------
#define NB   4
#define NS   1024
#define ND   1024          // model dim D
#define NE   8             // experts
#define NH   4096          // hidden = 4*D
#define NTOK 4096          // N = B*S
#define CAP  512           // capacity = N/E
#define TOPK 2

typedef __attribute__((ext_vector_type(16))) __bf16 v16bf;
typedef __attribute__((ext_vector_type(8)))  __bf16 v8bf;
typedef __attribute__((ext_vector_type(8)))  float  v8f;

// ---------------------------------------------------------------------------
// 1) Router: logits = x @ hash_kernel ; top-2 ; softmax over the 2 logits.
// ---------------------------------------------------------------------------
__global__ void moe_router_kernel(const float* __restrict__ x,
                                  const float* __restrict__ hk,
                                  int* __restrict__ eidx,
                                  float* __restrict__ gates) {
  const int lane = threadIdx.x & 31;
  const int wv   = threadIdx.x >> 5;
  const int tok  = blockIdx.x * 8 + wv;
  if (tok >= NTOK) return;

  float acc[NE];
#pragma unroll
  for (int e = 0; e < NE; ++e) acc[e] = 0.f;

  const float* xr = x + (size_t)tok * ND;
  for (int d = lane; d < ND; d += 32) {
    const float xv = xr[d];
    const float* hr = hk + (size_t)d * NE;
#pragma unroll
    for (int e = 0; e < NE; ++e) acc[e] += xv * hr[e];
  }
#pragma unroll
  for (int e = 0; e < NE; ++e) {
#pragma unroll
    for (int off = 16; off > 0; off >>= 1)
      acc[e] += __shfl_xor(acc[e], off, 32);
  }
  if (lane == 0) {
    int b0 = 0; float l0 = acc[0];
#pragma unroll
    for (int e = 1; e < NE; ++e) if (acc[e] > l0) { l0 = acc[e]; b0 = e; }
    int b1 = -1; float l1 = -INFINITY;
#pragma unroll
    for (int e = 0; e < NE; ++e) if (e != b0 && acc[e] > l1) { l1 = acc[e]; b1 = e; }
    const float t  = expf(l1 - l0);
    const float g1 = t / (1.f + t);
    const float g0 = 1.f / (1.f + t);
    eidx[2 * tok + 0]  = b0;  eidx[2 * tok + 1]  = b1;
    gates[2 * tok + 0] = g0;  gates[2 * tok + 1] = g1;
  }
}

// ---------------------------------------------------------------------------
// 2) Capacity scan: independent cumsum counter per (k, e) over token order.
// ---------------------------------------------------------------------------
__global__ void moe_priority_kernel(const int* __restrict__ eidx,
                                    int* __restrict__ slots,
                                    int* __restrict__ src) {
  const int tid = threadIdx.x;
  for (int i = tid; i < TOPK * NE * CAP; i += 256) src[i] = -1;
  __syncthreads();
  if (tid < TOPK) {
    const int k = tid;
    int cnt[NE];
#pragma unroll
    for (int e = 0; e < NE; ++e) cnt[e] = 0;
    for (int n = 0; n < NTOK; ++n) {
      const int e = eidx[2 * n + k];
      const int p = ++cnt[e];
      if (p <= CAP) {
        slots[2 * n + k] = p - 1;
        src[k * (NE * CAP) + e * CAP + (p - 1)] = n;
      } else {
        slots[2 * n + k] = -1;
      }
    }
  }
}

// ---------------------------------------------------------------------------
// 3) Transpose + fp32->bf16 of expert weights: in [R,C] -> out [C,R] per expert
// ---------------------------------------------------------------------------
__global__ void moe_transpose_cvt_kernel(const float* __restrict__ in,
                                         __bf16* __restrict__ out,
                                         int R, int Cin) {
  __shared__ float tile[32][33];
  const int e  = blockIdx.z;
  const float* ine = in + (size_t)e * R * Cin;
  __bf16*     oute = out + (size_t)e * R * Cin;
  const int c0 = blockIdx.x * 32, r0 = blockIdx.y * 32;
  const int tx = threadIdx.x, ty = threadIdx.y;
#pragma unroll
  for (int i = 0; i < 32; i += 8)
    tile[ty + i][tx] = ine[(size_t)(r0 + ty + i) * Cin + (c0 + tx)];
  __syncthreads();
#pragma unroll
  for (int i = 0; i < 32; i += 8)
    oute[(size_t)(c0 + ty + i) * R + (r0 + tx)] = (__bf16)tile[tx][ty + i];
}

// ---------------------------------------------------------------------------
// 4) Gather tokens into expert slots (summing k=0/k=1 collisions), bf16 out.
// ---------------------------------------------------------------------------
__global__ void moe_gather_kernel(const float* __restrict__ x,
                                  const int* __restrict__ src,
                                  __bf16* __restrict__ Xe) {
  const int r  = blockIdx.x;                 // r = e*CAP + c
  const int s0 = src[r];
  const int s1 = src[NE * CAP + r];
#pragma unroll
  for (int j = 0; j < ND / 256; ++j) {
    const int d = threadIdx.x + 256 * j;
    float v = 0.f;
    if (s0 >= 0) v += x[(size_t)s0 * ND + d];
    if (s1 >= 0) v += x[(size_t)s1 * ND + d];
    Xe[(size_t)r * ND + d] = (__bf16)v;
  }
}

// ---------------------------------------------------------------------------
// WMMA fragment loaders (layouts per CDNA5 ISA 7.12.2, wave32).
// ---------------------------------------------------------------------------
__device__ __forceinline__ v16bf load_frag_a(const __bf16* base, int ld,
                                             int row0, int k0, int lane) {
  const int m    = lane & 15;
  const int koff = (lane & 16) ? 8 : 0;
  const __bf16* p = base + (size_t)(row0 + m) * ld + k0 + koff;
  const v8bf lo = *(const v8bf*)p;
  const v8bf hi = *(const v8bf*)(p + 16);
  return __builtin_shufflevector(lo, hi, 0, 1, 2, 3, 4, 5, 6, 7,
                                 8, 9, 10, 11, 12, 13, 14, 15);
}

__device__ __forceinline__ v16bf load_frag_b(const __bf16* bt, int ld,
                                             int n0, int k0, int lane) {
  const int n    = lane & 15;
  const int koff = (lane & 16) ? 8 : 0;
  const __bf16* p = bt + (size_t)(n0 + n) * ld + k0 + koff;
  const v8bf lo = *(const v8bf*)p;
  const v8bf hi = *(const v8bf*)(p + 16);
  return __builtin_shufflevector(lo, hi, 0, 1, 2, 3, 4, 5, 6, 7,
                                 8, 9, 10, 11, 12, 13, 14, 15);
}

__device__ __forceinline__ float gelu_tanh(float v) {
  return 0.5f * v * (1.f + tanhf(0.7978845608028654f * (v + 0.044715f * v * v * v)));
}

// CDNA5 async global -> LDS copy, 16 bytes per lane, tracked by ASYNCcnt.
__device__ __forceinline__ void async_copy_b128(unsigned lds_addr,
                                                const __bf16* gaddr) {
  asm volatile("global_load_async_to_lds_b128 %0, %1, off"
               :: "v"(lds_addr), "v"(gaddr)
               : "memory");
}
__device__ __forceinline__ void wait_async0() {
  asm volatile("s_wait_asynccnt 0x0" ::: "memory");
}

// ---------------------------------------------------------------------------
// 5/6) bf16 WMMA GEMM with async-LDS double buffering.
//   Out[e] (MxNn) = A[e] (MxK) @ B, where BT[e] is the Nn x K transposed view.
//   Block = 256 threads (8 waves as 4 M-waves x 2 N-waves).
//   Block tile: 128(M) x 128(N), K-step 64. Wave tile: 32 x 64 (8 WMMA accs).
//   LDS: double-buffered A(128x64) + BT(128x64) bf16 tiles = 64 KB.
//   All 12 LDS fragment loads of a K-block are issued before any WMMA so the
//   matrix ops retire against partial s_wait_dscnt instead of full stalls.
// ---------------------------------------------------------------------------
#define TILE_AB_ELEMS  (128 * 64)           // per tile, per operand
#define BUF_ELEMS      (2 * TILE_AB_ELEMS)  // A tile + B tile

template <bool GELU_BF16OUT>
__global__ void moe_gemm_kernel(const __bf16* __restrict__ A,
                                const __bf16* __restrict__ BT,
                                void* __restrict__ Out,
                                int M, int Nn, int Kd) {
  __shared__ __bf16 smem[2][BUF_ELEMS];

  const int e    = blockIdx.z;
  const __bf16* Ae = A  + (size_t)e * M * Kd;
  const __bf16* Be = BT + (size_t)e * Nn * Kd;
  const int lane = threadIdx.x & 31;
  const int w    = threadIdx.x >> 5;
  const int wm   = w & 3;                   // M-wave 0..3
  const int wn   = w >> 2;                  // N-wave 0..1
  const int m0   = blockIdx.y * 128;        // block M origin
  const int n0   = blockIdx.x * 128;        // block N origin

  const unsigned lds_base = (unsigned)(uintptr_t)(&smem[0][0]);

  // Issue async copies of one 64-wide K-block of A and BT into buffer bsel.
  auto issue_copies = [&](int kblk, int bsel) {
    const unsigned lb = lds_base + (unsigned)bsel * (BUF_ELEMS * 2 /*bytes*/);
#pragma unroll
    for (int j = 0; j < 4; ++j) {
      const int c   = threadIdx.x + 256 * j;   // 16B chunk id, 0..1023
      const int row = c >> 3;                  // 8 chunks per 64-elem row
      const int c16 = c & 7;
      async_copy_b128(lb + (unsigned)c * 16,
                      Ae + (size_t)(m0 + row) * Kd + kblk + c16 * 8);
    }
#pragma unroll
    for (int j = 0; j < 4; ++j) {
      const int c   = threadIdx.x + 256 * j;
      const int row = c >> 3;
      const int c16 = c & 7;
      async_copy_b128(lb + (unsigned)(TILE_AB_ELEMS * 2) + (unsigned)c * 16,
                      Be + (size_t)(n0 + row) * Kd + kblk + c16 * 8);
    }
  };

  v8f acc0[4], acc1[4];
#pragma unroll
  for (int t = 0; t < 4; ++t) {
    acc0[t] = (v8f){0.f, 0.f, 0.f, 0.f, 0.f, 0.f, 0.f, 0.f};
    acc1[t] = (v8f){0.f, 0.f, 0.f, 0.f, 0.f, 0.f, 0.f, 0.f};
  }

  // Prologue: fill buffer 0.
  issue_copies(0, 0);
  wait_async0();
  __syncthreads();

  const int KB = Kd / 64;
  for (int kb = 0; kb < KB; ++kb) {
    const int cur = kb & 1;
    if (kb + 1 < KB) issue_copies((kb + 1) * 64, cur ^ 1);

    const __bf16* As = &smem[cur][0];
    const __bf16* Bs = &smem[cur][TILE_AB_ELEMS];

    // Batch all fragment loads for this K-64 block, then do all 16 WMMAs.
    v16bf af[2][2], bf[2][4];
#pragma unroll
    for (int s = 0; s < 2; ++s) {
      af[s][0] = load_frag_a(As, 64, wm * 32,      s * 32, lane);
      af[s][1] = load_frag_a(As, 64, wm * 32 + 16, s * 32, lane);
#pragma unroll
      for (int t = 0; t < 4; ++t)
        bf[s][t] = load_frag_b(Bs, 64, wn * 64 + 16 * t, s * 32, lane);
    }
#pragma unroll
    for (int s = 0; s < 2; ++s)
#pragma unroll
      for (int t = 0; t < 4; ++t) {
        acc0[t] = __builtin_amdgcn_wmma_f32_16x16x32_bf16(
            false, af[s][0], false, bf[s][t], (short)0, acc0[t], false, false);
        acc1[t] = __builtin_amdgcn_wmma_f32_16x16x32_bf16(
            false, af[s][1], false, bf[s][t], (short)0, acc1[t], false, false);
      }

    wait_async0();       // our prefetch into buf^1 landed
    __syncthreads();     // everyone done reading buf[cur]
  }

  // Store: C/D layout — lane holds column N = lane&15; rows M = (lane&16?8:0)+r
  const int nlo  = lane & 15;
  const int mofs = (lane & 16) ? 8 : 0;
  const int rbase0 = m0 + wm * 32 + mofs;
  const int cbase  = n0 + wn * 64 + nlo;
  if (GELU_BF16OUT) {
    __bf16* O = (__bf16*)Out + (size_t)e * M * Nn;
#pragma unroll
    for (int t = 0; t < 4; ++t)
#pragma unroll
      for (int r = 0; r < 8; ++r) {
        O[(size_t)(rbase0 + r) * Nn + (cbase + 16 * t)]      = (__bf16)gelu_tanh(acc0[t][r]);
        O[(size_t)(rbase0 + 16 + r) * Nn + (cbase + 16 * t)] = (__bf16)gelu_tanh(acc1[t][r]);
      }
  } else {
    float* O = (float*)Out + (size_t)e * M * Nn;
#pragma unroll
    for (int t = 0; t < 4; ++t)
#pragma unroll
      for (int r = 0; r < 8; ++r) {
        O[(size_t)(rbase0 + r) * Nn + (cbase + 16 * t)]      = acc0[t][r];
        O[(size_t)(rbase0 + 16 + r) * Nn + (cbase + 16 * t)] = acc1[t][r];
      }
  }
}

// ---------------------------------------------------------------------------
// 7) Combine: out[n] = g0 * Y[e0,c0] + g1 * Y[e1,c1]  (0 for dropped slots)
// ---------------------------------------------------------------------------
__global__ void moe_combine_kernel(const float* __restrict__ Y,
                                   const int* __restrict__ eidx,
                                   const int* __restrict__ slots,
                                   const float* __restrict__ gates,
                                   float* __restrict__ out) {
  const int n  = blockIdx.x;
  const int e0 = eidx[2 * n], e1 = eidx[2 * n + 1];
  const int s0 = slots[2 * n], s1 = slots[2 * n + 1];
  const float g0 = gates[2 * n], g1 = gates[2 * n + 1];
#pragma unroll
  for (int j = 0; j < ND / 256; ++j) {
    const int d = threadIdx.x + 256 * j;
    float v = 0.f;
    if (s0 >= 0) v += g0 * Y[((size_t)e0 * CAP + s0) * ND + d];
    if (s1 >= 0) v += g1 * Y[((size_t)e1 * CAP + s1) * ND + d];
    out[(size_t)n * ND + d] = v;
  }
}

// ---------------------------------------------------------------------------
extern "C" void kernel_launch(void* const* d_in, const int* in_sizes, int n_in,
                              void* d_out, int out_size, void* d_ws, size_t ws_size,
                              hipStream_t stream) {
  (void)in_sizes; (void)n_in; (void)out_size; (void)ws_size;
  const float* x   = (const float*)d_in[0];   // [N, D]
  const float* hk  = (const float*)d_in[1];   // [D, E]
  const float* w1  = (const float*)d_in[2];   // [E, D, 4D]
  const float* w2  = (const float*)d_in[3];   // [E, 4D, D]
  float* out = (float*)d_out;                 // [N, D]

  char* ws = (char*)d_ws;
  size_t off = 0;
  auto take = [&](size_t bytes) -> char* {
    char* p = ws + off;
    off = (off + bytes + 255) & ~(size_t)255;
    return p;
  };
  int*    eidx  = (int*)   take(sizeof(int)   * NTOK * TOPK);
  float*  gates = (float*) take(sizeof(float) * NTOK * TOPK);
  int*    slots = (int*)   take(sizeof(int)   * NTOK * TOPK);
  int*    src   = (int*)   take(sizeof(int)   * TOPK * NE * CAP);
  __bf16* w1t   = (__bf16*)take(sizeof(__bf16) * (size_t)NE * NH * ND);  // [E][4D][D]
  __bf16* w2t   = (__bf16*)take(sizeof(__bf16) * (size_t)NE * ND * NH);  // [E][D][4D]
  __bf16* xe    = (__bf16*)take(sizeof(__bf16) * (size_t)NE * CAP * ND); // [E][C][D]
  __bf16* hbuf  = (__bf16*)take(sizeof(__bf16) * (size_t)NE * CAP * NH); // [E][C][4D]
  float*  ybuf  = (float*) take(sizeof(float)  * (size_t)NE * CAP * ND); // [E][C][D]

  moe_router_kernel<<<NTOK / 8, 256, 0, stream>>>(x, hk, eidx, gates);
  moe_priority_kernel<<<1, 256, 0, stream>>>(eidx, slots, src);
  moe_transpose_cvt_kernel<<<dim3(NH / 32, ND / 32, NE), dim3(32, 8), 0, stream>>>(
      w1, w1t, ND, NH);
  moe_transpose_cvt_kernel<<<dim3(ND / 32, NH / 32, NE), dim3(32, 8), 0, stream>>>(
      w2, w2t, NH, ND);
  moe_gather_kernel<<<NE * CAP, 256, 0, stream>>>(x, src, xe);
  // H = gelu(Xe @ W1): M=512, N=4096, K=1024
  moe_gemm_kernel<true><<<dim3(NH / 128, CAP / 128, NE), 256, 0, stream>>>(
      xe, w1t, (void*)hbuf, CAP, NH, ND);
  // Y = H @ W2:        M=512, N=1024, K=4096
  moe_gemm_kernel<false><<<dim3(ND / 128, CAP / 128, NE), 256, 0, stream>>>(
      hbuf, w2t, (void*)ybuf, CAP, ND, NH);
  moe_combine_kernel<<<NTOK, 256, 0, stream>>>(ybuf, eidx, slots, gates, out);
}
